// MultiHeadAttention_64862596104701
// MI455X (gfx1250) — matrix operand order repaired and checked by reading, not compiled, not run on hardware
//
#include <hip/hip_runtime.h>

#define DEVINL __device__ __forceinline__

typedef __attribute__((ext_vector_type(16))) __bf16 bf16x16;
typedef __attribute__((ext_vector_type(8)))  __bf16 bf16x8;
typedef __attribute__((ext_vector_type(8)))  float  f32x8;
typedef __attribute__((ext_vector_type(4)))  unsigned int uint4v;
typedef __attribute__((ext_vector_type(8)))  int int8v;
typedef __attribute__((ext_vector_type(4)))  int int4v;

constexpr int Bb = 2;      // batch
constexpr int S  = 2048;   // sequence
constexpr int H  = 12;     // heads
constexpr int DK = 64;     // head dim
constexpr int DM = 768;    // model dim

// ---------------------------------------------------------------------------
// TDM availability (this toolchain: 6-arg builtin, clang-23/therock headers;
// host pass sees 0)
// ---------------------------------------------------------------------------
#if defined(__has_builtin)
#if __has_builtin(__builtin_amdgcn_tensor_load_to_lds) && \
    __has_builtin(__builtin_amdgcn_s_wait_tensorcnt)
#define USE_TDM 1
#endif
#endif
#ifndef USE_TDM
#define USE_TDM 0
#endif

#if USE_TDM
// Issue a 2D TDM tile load (global -> LDS) with optional LDS row padding.
// Dims/strides in data_size units; data_size_code: 0=1B,1=2B,2=4B,3=8B.
// pad: after 2^(interval_code+1) DWORDs stored, skip (amount_code+1) DWORDs.
DEVINL void tdm_load_2d(unsigned lds_addr, unsigned long long gaddr,
                        unsigned tile_d0, unsigned tile_d1,
                        unsigned tensor_d0, unsigned tensor_d1,
                        unsigned stride0, unsigned data_size_code,
                        unsigned pad_en, unsigned pad_interval_code,
                        unsigned pad_amount_code)
{
  // D# group 0: count=1 | lds_addr | global_addr(57b) | type=2 ("image")
  uint4v g0 = { 1u,
                lds_addr,
                (unsigned)gaddr,
                (unsigned)((gaddr >> 32) & 0x01FFFFFFull) | (2u << 30) };
  // D# group 1: wg_mask=0, data_size, pad fields, tensor/tile dims, stride0
  int8v g1 = {
    (int)((data_size_code << 16) | (pad_en << 20) |
          (pad_interval_code << 22) | (pad_amount_code << 25)),
    (int)((tensor_d0 & 0xFFFFu) << 16),                              // dim0 lo
    (int)(((tensor_d0 >> 16) & 0xFFFFu) | ((tensor_d1 & 0xFFFFu) << 16)),
    (int)(((tensor_d1 >> 16) & 0xFFFFu) | ((tile_d0 & 0xFFFFu) << 16)),
    (int)(tile_d1 & 0xFFFFu),                                        // tile_dim2=0
    (int)stride0,                                                    // stride0 lo32
    0, 0 };                                                          // stride0 hi / stride1
  int4v gz4 = { 0, 0, 0, 0 };              // groups 2/3: higher dims unused
  int8v gz8 = { 0, 0, 0, 0, 0, 0, 0, 0 };  // extra group (6-arg form): unused
  __builtin_amdgcn_tensor_load_to_lds(g0, g1, gz4, gz4, gz8, 0);
}
#endif

// ---------------------------------------------------------------------------
// WMMA helper: D = A(16x32 bf16) * B(32x16 bf16) + C(16x16 f32)
// ---------------------------------------------------------------------------
DEVINL f32x8 wmma_bf16(bf16x16 a, bf16x16 b, f32x8 c) {
  return __builtin_amdgcn_wmma_f32_16x16x32_bf16(
      /*neg_a=*/false, a, /*neg_b=*/false, b,
      /*c_mod=*/(short)0, c, /*reuse_a=*/false, /*reuse_b=*/false);
}

// A-matrix fragment (16x32, M x K), per ISA 7.12.2:
// lane L: row M = L&15; K = kb+{0..7} then kb+16+{0..7}, kb = (L>>4)*8
DEVINL bf16x16 load_fragA(const __bf16* base, int row_stride) {
  const int lane = threadIdx.x & 31;
  const __bf16* p = base + (lane & 15) * row_stride + ((lane >> 4) << 3);
  bf16x8 lo = *(const bf16x8*)(p);
  bf16x8 hi = *(const bf16x8*)(p + 16);
  return __builtin_shufflevector(lo, hi, 0,1,2,3,4,5,6,7,8,9,10,11,12,13,14,15);
}

// B-matrix fragment (32x16, K x N) from column-major storage (row = N, K contig):
// lane L: col N = L&15; K = kb+{0..7} then kb+16+{0..7}, kb = (L>>4)*8
DEVINL bf16x16 load_fragB(const __bf16* base, int row_stride) {
  const int lane = threadIdx.x & 31;
  const __bf16* p = base + (lane & 15) * row_stride + ((lane >> 4) << 3);
  bf16x8 lo = *(const bf16x8*)(p);
  bf16x8 hi = *(const bf16x8*)(p + 16);
  return __builtin_shufflevector(lo, hi, 0,1,2,3,4,5,6,7,8,9,10,11,12,13,14,15);
}

// ---------------------------------------------------------------------------
// Kernel 1: per-head QKV projection.  out[b][h][s][n] = sum_k x[b][s][k]*W[h][k][n]
// Grid: (B*S/128, H, 3) ; block 256 (8 waves).  f32 in -> bf16 out (workspace).
// (f32->bf16 conversion during staging; TDM cannot convert, so manual copies.)
// ---------------------------------------------------------------------------
__global__ __launch_bounds__(256)
void proj_kernel(const float* __restrict__ xq, const float* __restrict__ xk,
                 const float* __restrict__ xv,
                 const float* __restrict__ Wq, const float* __restrict__ Wk,
                 const float* __restrict__ Wv,
                 __bf16* __restrict__ qh, __bf16* __restrict__ kh,
                 __bf16* __restrict__ vh)
{
  const int which = blockIdx.z;
  const float* x = (which == 0) ? xq : (which == 1) ? xk : xv;
  const float* W = (which == 0) ? Wq : (which == 1) ? Wk : Wv;
  __bf16* out    = (which == 0) ? qh : (which == 1) ? kh : vh;

  const int h  = blockIdx.y;
  const int g0 = blockIdx.x * 128;          // row in [0, B*S)

  __shared__ __align__(16) __bf16 Alds[128 * 40];   // 128 x 32, stride 40
  __shared__ __align__(16) __bf16 Blds[64 * 40];    // transposed: [n][k], 64 x 32

  const int t = threadIdx.x, wid = t >> 5, lane = t & 31;

  f32x8 acc[4];
#pragma unroll
  for (int i = 0; i < 4; ++i)
#pragma unroll
    for (int j = 0; j < 8; ++j) acc[i][j] = 0.0f;

  for (int k0 = 0; k0 < DM; k0 += 32) {
    // stage A tile: 128x32 f32 -> bf16
    {
      const int col = t & 31, rb = t >> 5;
#pragma unroll
      for (int i = 0; i < 16; ++i) {
        const int row = rb + i * 8;
        Alds[row * 40 + col] = (__bf16)x[(size_t)(g0 + row) * DM + k0 + col];
      }
    }
    // stage B tile transposed: Blds[n][k] = W[h][k0+k][n]
    {
      const int n = t & 63, kb = t >> 6;
#pragma unroll
      for (int i = 0; i < 8; ++i) {
        const int k = kb + i * 4;
        Blds[n * 40 + k] = (__bf16)W[((size_t)h * DM + (k0 + k)) * DK + n];
      }
    }
    __syncthreads();
    const bf16x16 af = load_fragA(&Alds[wid * 16 * 40], 40);
#pragma unroll
    for (int nt = 0; nt < 4; ++nt) {
      const bf16x16 bf = load_fragB(&Blds[nt * 16 * 40], 40);
      acc[nt] = wmma_bf16(af, bf, acc[nt]);
    }
    __syncthreads();
  }

  const int b     = g0 / S;
  const int srow0 = (g0 % S) + wid * 16;
#pragma unroll
  for (int nt = 0; nt < 4; ++nt) {
    const int n = nt * 16 + (lane & 15);
#pragma unroll
    for (int r = 0; r < 8; ++r) {
      const int m = (lane < 16) ? r : r + 8;
      out[(((size_t)b * H + h) * S + (srow0 + m)) * DK + n] = (__bf16)acc[nt][r];
    }
  }
}

// ---------------------------------------------------------------------------
// Kernel 2: flash attention per (b,h).  128 query rows per WG, 64-key blocks.
// Grid: (B*S/128, H) ; block 256 (8 waves, each owns 16 query rows).
// K tile staged by the Tensor Data Mover (row pad 128B->144B via D# pad fields).
// ---------------------------------------------------------------------------
__global__ __launch_bounds__(256)
void flash_kernel(const __bf16* __restrict__ qh, const __bf16* __restrict__ kh,
                  const __bf16* __restrict__ vh, const int* __restrict__ mask,
                  __bf16* __restrict__ attn /* [B*S, DM] bf16 */)
{
  const int h  = blockIdx.y;
  const int qb = blockIdx.x;
  const int b  = qb / (S / 128);
  const int q0 = (qb % (S / 128)) * 128;

  __shared__ __align__(16) __bf16 Klds[64 * 72];       // [key][dim]
  __shared__ __align__(16) __bf16 Vlds[64 * 72];       // transposed: [dim][key]
  __shared__ __align__(16) __bf16 Plds[8][16 * 72];    // per-wave P strip

  const int t = threadIdx.x, wid = t >> 5, lane = t & 31;
  const size_t headbase = (((size_t)b * H + h) * S) * DK;

  // preload Q fragments (A layout) straight from global
  bf16x16 qfrag[2];
  {
    const int row = q0 + wid * 16 + (lane & 15);
    const __bf16* qrow = qh + headbase + (size_t)row * DK;
    const int kb = (lane >> 4) << 3;
#pragma unroll
    for (int f = 0; f < 2; ++f) {
      const __bf16* p = qrow + f * 32 + kb;
      bf16x8 lo = *(const bf16x8*)(p);
      bf16x8 hi = *(const bf16x8*)(p + 16);
      qfrag[f] = __builtin_shufflevector(lo, hi, 0,1,2,3,4,5,6,7,8,9,10,11,12,13,14,15);
    }
  }

  f32x8 Oacc[4];
  float mrun[8], lrun[8];
#pragma unroll
  for (int i = 0; i < 4; ++i)
#pragma unroll
    for (int j = 0; j < 8; ++j) Oacc[i][j] = 0.0f;
#pragma unroll
  for (int r = 0; r < 8; ++r) { mrun[r] = -3.0e38f; lrun[r] = 0.0f; }

  const float scale = 0.125f;   // 1/sqrt(64)

  for (int kb0 = 0; kb0 < S; kb0 += 64) {
#if USE_TDM
    // TDM: DMA K tile (64 rows x 128B, pad each 32 DWORDs by 4 DWORDs -> 144B rows)
    if (wid == 0) {
      tdm_load_2d((unsigned)(uintptr_t)&Klds[0],
                  (unsigned long long)(uintptr_t)(kh + headbase + (size_t)kb0 * DK),
                  /*tile_d0=*/DK, /*tile_d1=*/64,
                  /*tensor_d0=*/DK, /*tensor_d1=*/S,
                  /*stride0=*/DK, /*data_size=2B*/1u,
                  /*pad_en=*/1u, /*interval: 32 dw*/4u, /*amount: 4 dw*/3u);
    }
#else
    {
      const int key = t >> 2, dc = (t & 3) * 16;
      const __bf16* src = kh + headbase + (size_t)(kb0 + key) * DK + dc;
      *(bf16x8*)&Klds[key * 72 + dc]     = *(const bf16x8*)(src);
      *(bf16x8*)&Klds[key * 72 + dc + 8] = *(const bf16x8*)(src + 8);
    }
#endif
    // stage V block transposed: Vlds[dim][key]  (TDM cannot transpose)
    {
      const int dim = t & 63, kc = t >> 6;
#pragma unroll
      for (int i = 0; i < 16; ++i) {
        const int key = kc + i * 4;
        Vlds[dim * 72 + key] = vh[headbase + (size_t)(kb0 + key) * DK + dim];
      }
    }
#if USE_TDM
    __builtin_amdgcn_s_wait_tensorcnt(0);   // no-op for non-issuing waves
#endif
    __syncthreads();

    if (kb0 + 64 < S)   // prefetch next K block -> global_prefetch_b8
      __builtin_prefetch(kh + headbase + (size_t)(kb0 + 64) * DK + (t & 63), 0, 1);

    // scores: S = Q(16x64) * K^T  (K rows of Klds are the B "columns")
    f32x8 st[4];
#pragma unroll
    for (int i = 0; i < 4; ++i)
#pragma unroll
      for (int j = 0; j < 8; ++j) st[i][j] = 0.0f;
#pragma unroll
    for (int nt = 0; nt < 4; ++nt)
#pragma unroll
      for (int f = 0; f < 2; ++f) {
        const bf16x16 kf = load_fragB(&Klds[nt * 16 * 72 + f * 32], 72);
        st[nt] = wmma_bf16(qfrag[f], kf, st[nt]);
      }

    // scale + mask
#pragma unroll
    for (int nt = 0; nt < 4; ++nt) {
      const int col = kb0 + nt * 16 + (lane & 15);
#pragma unroll
      for (int r = 0; r < 8; ++r) {
        const int qq = q0 + wid * 16 + ((lane < 16) ? r : r + 8);
        const int mv = mask[((size_t)b * S + qq) * S + col];
        st[nt][r] = (mv == 0) ? -1e9f : st[nt][r] * scale;
      }
    }

    // online softmax (row = M; lanes 0..15 / 16..31 each hold a full row)
#pragma unroll
    for (int r = 0; r < 8; ++r) {
      float mx = fmaxf(fmaxf(st[0][r], st[1][r]), fmaxf(st[2][r], st[3][r]));
      mx = fmaxf(mx, __shfl_xor(mx, 1));
      mx = fmaxf(mx, __shfl_xor(mx, 2));
      mx = fmaxf(mx, __shfl_xor(mx, 4));
      mx = fmaxf(mx, __shfl_xor(mx, 8));
      const float mnew  = fmaxf(mrun[r], mx);
      const float alpha = __expf(mrun[r] - mnew);
      mrun[r] = mnew;
      float rs = 0.0f;
#pragma unroll
      for (int nt = 0; nt < 4; ++nt) {
        const float p = __expf(st[nt][r] - mnew);
        st[nt][r] = p;
        rs += p;
      }
      rs += __shfl_xor(rs, 1); rs += __shfl_xor(rs, 2);
      rs += __shfl_xor(rs, 4); rs += __shfl_xor(rs, 8);
      lrun[r] = lrun[r] * alpha + rs;
#pragma unroll
      for (int nt = 0; nt < 4; ++nt) Oacc[nt][r] *= alpha;
    }

    // spill P (bf16) to this wave's LDS strip, reread as A fragments
    __bf16* pw = &Plds[wid][0];
#pragma unroll
    for (int nt = 0; nt < 4; ++nt) {
      const int key = nt * 16 + (lane & 15);
#pragma unroll
      for (int r = 0; r < 8; ++r) {
        const int m = (lane < 16) ? r : r + 8;
        pw[m * 72 + key] = (__bf16)st[nt][r];
      }
    }

    // O += P(16x64) * V(64x64)
#pragma unroll
    for (int nt = 0; nt < 4; ++nt)
#pragma unroll
      for (int f = 0; f < 2; ++f) {
        const bf16x16 pf = load_fragA(pw + f * 32, 72);
        const bf16x16 vf = load_fragB(&Vlds[nt * 16 * 72 + f * 32], 72);
        Oacc[nt] = wmma_bf16(pf, vf, Oacc[nt]);
      }
    __syncthreads();
  }

  // epilogue: normalize, store concat-head layout [B*S, H*DK] bf16
#pragma unroll
  for (int nt = 0; nt < 4; ++nt) {
    const int n = nt * 16 + (lane & 15);
#pragma unroll
    for (int r = 0; r < 8; ++r) {
      const int m = (lane < 16) ? r : r + 8;
      const int srow = q0 + wid * 16 + m;
      attn[((size_t)b * S + srow) * DM + h * DK + n] = (__bf16)(Oacc[nt][r] / lrun[r]);
    }
  }
}

// ---------------------------------------------------------------------------
// Kernel 3: output projection  out[M,768] = attn[M,768](bf16) x Wo[768,768](f32)
// Grid: (B*S/128, DM/64) ; block 256.  A tile staged by TDM (bf16 copy,
// 64B rows padded to 80B -> LDS stride 40 elements).
// ---------------------------------------------------------------------------
__global__ __launch_bounds__(256)
void outproj_kernel(const __bf16* __restrict__ attn, const float* __restrict__ Wo,
                    float* __restrict__ out)
{
  const int g0 = blockIdx.x * 128;
  const int n0 = blockIdx.y * 64;

  __shared__ __align__(16) __bf16 Alds[128 * 40];
  __shared__ __align__(16) __bf16 Blds[64 * 40];

  const int t = threadIdx.x, wid = t >> 5, lane = t & 31;

  f32x8 acc[4];
#pragma unroll
  for (int i = 0; i < 4; ++i)
#pragma unroll
    for (int j = 0; j < 8; ++j) acc[i][j] = 0.0f;

  for (int k0 = 0; k0 < DM; k0 += 32) {
#if USE_TDM
    // TDM: A tile 128 rows x 64B, pad each 16 DWORDs by 4 DWORDs -> 80B rows
    if (wid == 0) {
      tdm_load_2d((unsigned)(uintptr_t)&Alds[0],
                  (unsigned long long)(uintptr_t)(attn + (size_t)g0 * DM + k0),
                  /*tile_d0=*/32, /*tile_d1=*/128,
                  /*tensor_d0=*/DM, /*tensor_d1=*/Bb * S,
                  /*stride0=*/DM, /*data_size=2B*/1u,
                  /*pad_en=*/1u, /*interval: 16 dw*/3u, /*amount: 4 dw*/3u);
    }
#else
    {
      const int col = t & 31, rb = t >> 5;
#pragma unroll
      for (int i = 0; i < 16; ++i) {
        const int row = rb + i * 8;
        Alds[row * 40 + col] = attn[(size_t)(g0 + row) * DM + k0 + col];
      }
    }
#endif
    // stage B tile transposed: Blds[n][k] = Wo[k0+k][n0+n] (f32 -> bf16)
    {
      const int n = t & 63, kb = t >> 6;
#pragma unroll
      for (int i = 0; i < 8; ++i) {
        const int k = kb + i * 4;
        Blds[n * 40 + k] = (__bf16)Wo[(size_t)(k0 + k) * DM + n0 + n];
      }
    }
#if USE_TDM
    __builtin_amdgcn_s_wait_tensorcnt(0);
#endif
    __syncthreads();
    const bf16x16 af = load_fragA(&Alds[wid * 16 * 40], 40);
#pragma unroll
    for (int nt = 0; nt < 4; ++nt) {
      const bf16x16 bf = load_fragB(&Blds[nt * 16 * 40], 40);
      acc[nt] = wmma_bf16(af, bf, acc[nt]);
    }
    __syncthreads();
  }

#pragma unroll
  for (int nt = 0; nt < 4; ++nt) {
    const int n = n0 + nt * 16 + (lane & 15);
#pragma unroll
    for (int r = 0; r < 8; ++r) {
      const int m = (lane < 16) ? r : r + 8;
      out[(size_t)(g0 + wid * 16 + m) * DM + n] = acc[nt][r];
    }
  }
}

// ---------------------------------------------------------------------------
extern "C" void kernel_launch(void* const* d_in, const int* in_sizes, int n_in,
                              void* d_out, int out_size, void* d_ws, size_t ws_size,
                              hipStream_t stream) {
  const float* q    = (const float*)d_in[0];
  const float* k    = (const float*)d_in[1];
  const float* v    = (const float*)d_in[2];
  const int*   mask = (const int*)  d_in[3];
  const float* Wq   = (const float*)d_in[4];
  const float* Wk   = (const float*)d_in[5];
  const float* Wv   = (const float*)d_in[6];
  const float* Wo   = (const float*)d_in[7];
  float* out = (float*)d_out;

  // workspace carve-up (bf16): qh | kh | vh | attn  (~25 MB total)
  __bf16* ws = (__bf16*)d_ws;
  const size_t headElems = (size_t)Bb * H * S * DK;   // 3,145,728
  __bf16* qh   = ws;
  __bf16* kh   = qh + headElems;
  __bf16* vh   = kh + headElems;
  __bf16* attn = vh + headElems;                       // B*S*DM elems

  proj_kernel<<<dim3(Bb * S / 128, H, 3), 256, 0, stream>>>(
      q, k, v, Wq, Wk, Wv, qh, kh, vh);
  flash_kernel<<<dim3(Bb * S / 128, H), 256, 0, stream>>>(
      qh, kh, vh, mask, attn);
  outproj_kernel<<<dim3(Bb * S / 128, DM / 64), 256, 0, stream>>>(
      attn, Wo, out);
}
